// GateAttentionUnit_6846177870012
// MI455X (gfx1250) — matrix operand order repaired and checked by reading, not compiled, not run on hardware
//
#include <hip/hip_runtime.h>
#include <hip/hip_bf16.h>

// GAU: B=32768, S=50, UNIT=46, KEY=16. One block per batch, 256 threads (8 wave32).
// All 4 matmuls via v_wmma_f32_16x16x32_f16 on padded tiles held in LDS.
// x activations staged via gfx1250 GLOBAL_LOAD_ASYNC_TO_LDS (ASYNCcnt) path.

#define SEQ     50
#define UNIT    46
#define KEY     16
#define SPAD    64   // padded S
#define UPAD    64   // padded K-dim for UNIT
#define OPAD    64   // padded output channels of GEMM1 (62 -> 64)
#define NEG_BIG 1000000000000.0f

typedef __attribute__((ext_vector_type(16))) _Float16 v16h;
typedef __attribute__((ext_vector_type(8)))  _Float16 v8h;
typedef __attribute__((ext_vector_type(8)))  float    v8f;

union V16U { v16h v; v8h h[2]; };

// ---- LDS layout (byte offsets, all 16-aligned, with phase-overlapped reuse) ----
// [0      .. 8192 )  xh   : 64x64 f16  (phase 0-1)      \ overlaid by
// [8192   .. 16384)  wih  : 64x64 f16  (phase 0-1)      /  Af32: 64x64 f32 (phase 2-S)
// [16384  .. 22528)  vT   : 48x64 f16  (phase 1-3)   B-matrix of P*V (V transposed)
// [22528  .. 26624)  qh   : 64x32 f16  (phase 1-2)
// [26624  .. 30720)  kh   : 64x32 f16  (phase 1-2)
// [30720  .. 38912)  ph   : 64x64 f16  (phase S-3)   softmaxed probabilities
// [38912  .. 47104)  o1h  : 64x64 f16  (phase 3-4)
// [47104  .. 53248)  woh  : 48x64 f16  (phase 0-4)
// [53248  .. 62848)  outf : 50x48 f32  (phase 4-5); f32 x staging during phase 0
#define XH_OFF    0
#define WIH_OFF   8192
#define AF32_OFF  0
#define VT_OFF    16384
#define QH_OFF    22528
#define KH_OFF    26624
#define PH_OFF    30720
#define O1_OFF    38912
#define WOH_OFF   47104
#define OUTF_OFF  53248
#define BUF_BYTES 62848

__device__ inline v8f wmma_f16(v16h a, v16h b, v8f c) {
  // (neg_a, A, neg_b, B, c_mod, C, reuse_a, reuse_b)
  return __builtin_amdgcn_wmma_f32_16x16x32_f16(false, a, false, b, (short)0, c,
                                                false, false);
}

__device__ inline float fast_rcp(float v) { return __builtin_amdgcn_rcpf(v); }
__device__ inline float silu(float h) { return h * fast_rcp(1.f + __expf(-h)); }

// A-frag 16x32 f16: lane(0-15)=row m, k chunks [kb,kb+8) and [kb+16,kb+24), kb = 8*(lane>>4)
__device__ inline v16h load_a16(const _Float16* base, int row0, int k0, int stride, int lane) {
  int m  = lane & 15;
  int kb = (lane >> 4) << 3;
  const _Float16* p = base + (row0 + m) * stride + k0 + kb;
  V16U u;
  u.h[0] = *(const v8h*)(p);
  u.h[1] = *(const v8h*)(p + 16);
  return u.v;
}

// B-frag 32x16 f16: lane holds column n = n0+(lane&15), k chunks [kb,kb+8) and
// [kb+16,kb+24), kb = k0 + 8*(lane>>4). "base" stores B columns as contiguous rows.
__device__ inline v16h load_b16(const _Float16* base, int n0, int k0, int stride, int lane) {
  int n  = n0 + (lane & 15);
  int kb = k0 + ((lane >> 4) << 3);
  const _Float16* p = base + n * stride + kb;
  V16U u;
  u.h[0] = *(const v8h*)(p);
  u.h[1] = *(const v8h*)(p + 16);
  return u.v;
}

__global__ __launch_bounds__(256)
void gau_kernel(const float* __restrict__ x,      // (B,50,46)
                const float* __restrict__ mask,   // (B,50)
                const float* __restrict__ Wi,     // (108,46)
                const float* __restrict__ bi,     // (108)
                const float* __restrict__ Wo,     // (46,46)
                const float* __restrict__ bo,     // (46)
                const float* __restrict__ scale_q,
                const float* __restrict__ offset_q,
                const float* __restrict__ scale_k,
                const float* __restrict__ offset_k,
                const float* __restrict__ Wq,     // (1,46)
                const float* __restrict__ bq,     // (1)
                float* __restrict__ out)          // (B,46)
{
  __shared__ __align__(16) unsigned char buf[BUF_BYTES];
  __shared__ float sProb[64];
  __shared__ float sRed[2];

  _Float16* xh  = (_Float16*)(buf + XH_OFF);
  _Float16* wih = (_Float16*)(buf + WIH_OFF);
  float*    Af  = (float*)   (buf + AF32_OFF);
  _Float16* vT  = (_Float16*)(buf + VT_OFF);
  _Float16* qh  = (_Float16*)(buf + QH_OFF);
  _Float16* kh  = (_Float16*)(buf + KH_OFF);
  _Float16* ph  = (_Float16*)(buf + PH_OFF);
  _Float16* o1h = (_Float16*)(buf + O1_OFF);
  _Float16* woh = (_Float16*)(buf + WOH_OFF);
  float*    outf= (float*)   (buf + OUTF_OFF);
  float*    xs  = (float*)   (buf + OUTF_OFF);   // phase-0 staging (9200B <= 9600B)

  const int b    = blockIdx.x;
  const int tid  = threadIdx.x;
  const int lane = tid & 31;
  const int wave = tid >> 5;
  const size_t xbase = (size_t)b * SEQ * UNIT;

  // ---------------- Phase 0a: async-DMA this batch's activations into LDS -----
  // gfx1250 GLOBAL_LOAD_ASYNC_TO_LDS_B32: LDS dest addr in VGPR, no VGPR data.
  for (int i = tid; i < SEQ * UNIT; i += 256) {
    const float* gp = x + xbase + i;
    unsigned     lp = (unsigned)(unsigned long long)(xs + i);  // wave LDS offset
    asm volatile("global_load_async_to_lds_b32 %0, %1, off"
                 :: "v"(lp), "v"(gp) : "memory");
  }

  // ---------------- Phase 0b: stage weights + zero padded regions -------------
  for (int i = tid; i < OPAD * UPAD; i += 256) {     // Wi rows 46..107 (v,qk only)
    int o = i >> 6, u = i & 63;
    float v = (o < 62 && u < UNIT) ? Wi[(46 + o) * UNIT + u] : 0.f;
    wih[i] = (_Float16)v;
  }
  for (int i = tid; i < 48 * UPAD; i += 256) {
    int o = i >> 6, u = i & 63;
    float v = (o < UNIT && u < UNIT) ? Wo[o * UNIT + u] : 0.f;
    woh[i] = (_Float16)v;
  }
  for (int i = tid; i < 48 * SPAD; i += 256) vT[i]  = (_Float16)0.f;
  for (int i = tid; i < SPAD * 32; i += 256) { qh[i] = (_Float16)0.f; kh[i] = (_Float16)0.f; }
  for (int i = tid; i < SPAD * OPAD; i += 256) o1h[i] = (_Float16)0.f;

  // Per-wave fence on async DMA, then block-wide barrier before consuming xs.
  asm volatile("s_wait_asynccnt 0x0" ::: "memory");
  __syncthreads();

  // ---------------- Phase 0c: convert staged x -> padded f16 tile -------------
  for (int i = tid; i < SPAD * UPAD; i += 256) {
    int s = i >> 6, u = i & 63;
    float v = (s < SEQ && u < UNIT) ? xs[s * UNIT + u] : 0.f;
    xh[i] = (_Float16)v;
  }
  __syncthreads();

  // ---------------- Phase 1: H = silu(X @ Wi_sub^T + bi); emit vT, RoPE(q,k) ---
  for (int t = wave; t < 16; t += 8) {
    int mt = (t >> 2) << 4, nt = (t & 3) << 4;
    v8f acc = {};
#pragma unroll
    for (int k0 = 0; k0 < UPAD; k0 += 32)
      acc = wmma_f16(load_a16(xh, mt, k0, UPAD, lane),
                     load_b16(wih, nt, k0, UPAD, lane), acc);
    int np = nt + (lane & 15);               // output channel 0..63 (62 valid)
    int mb = mt + ((lane >> 4) << 3);
    if (np < 62) {
      float biv = bi[46 + np];
      if (np < UNIT) {                       // v channel -> store transposed
#pragma unroll
        for (int r = 0; r < 8; ++r) {
          float h = silu(acc[r] + biv);
          vT[np * SPAD + (mb + r)] = (_Float16)h;
        }
      } else {                               // qk channel -> scale/offset + RoPE
        int c = np - 46;                     // 0..15
        float sq = scale_q[c], oq = offset_q[c];
        float sk = scale_k[c], ok = offset_k[c];
        float div = __expf(-(float)(2 * (c >> 1)) * 0.5756462732485115f); // 1e4^(-2i/16)
#pragma unroll
        for (int r = 0; r < 8; ++r) {
          int s = mb + r;
          float h = silu(acc[r] + biv);
          float yq = h * sq + oq;
          float yk = h * sk + ok;
          float pq = __shfl_xor(yq, 1, 32);  // partner channel c^1 (same branch)
          float pk = __shfl_xor(yk, 1, 32);
          float ang = (float)s * div;
          float cs = __cosf(ang), sn = __sinf(ang);
          float qv = (c & 1) ? (yq * cs + pq * sn) : (yq * cs - pq * sn);
          float kv = (c & 1) ? (yk * cs + pk * sn) : (yk * cs - pk * sn);
          qh[s * 32 + c] = (_Float16)qv;
          kh[s * 32 + c] = (_Float16)kv;
        }
      }
    }
  }
  __syncthreads();

  // ---------------- Phase 2: logits = (Q K^T)/4, masked -> Af32 ----------------
  for (int t = wave; t < 16; t += 8) {
    int mt = (t >> 2) << 4, nt = (t & 3) << 4;
    v8f acc = {};
    acc = wmma_f16(load_a16(qh, mt, 0, 32, lane),
                   load_b16(kh, nt, 0, 32, lane), acc);
    int j  = nt + (lane & 15);
    int mb = mt + ((lane >> 4) << 3);
    float mj  = (j < SEQ) ? mask[(size_t)b * SEQ + j] : 0.f;
    float neg = (1.f - mj) * NEG_BIG;
#pragma unroll
    for (int r = 0; r < 8; ++r)
      Af[(mb + r) * SPAD + j] = acc[r] * 0.25f * mj - neg;
  }
  __syncthreads();

  // ---------------- Softmax over j (per row), write P as f16 ------------------
  if (tid < SPAD) {
    float* row = Af + tid * SPAD;
    float mx = -3.4e38f;
    for (int j = 0; j < SEQ; ++j) mx = fmaxf(mx, row[j]);
    float sum = 0.f;
    for (int j = 0; j < SEQ; ++j) sum += __expf(row[j] - mx);
    float inv = fast_rcp(sum);
    _Float16* pr = ph + tid * SPAD;
    for (int j = 0; j < SEQ; ++j) pr[j] = (_Float16)(__expf(row[j] - mx) * inv);
    for (int j = SEQ; j < SPAD; ++j) pr[j] = (_Float16)0.f;   // K-padding
  }
  __syncthreads();

  // ---------------- Phase 3: O1 = P @ V  (B = vT rows) ------------------------
  for (int t = wave; t < 12; t += 8) {
    int mt = (t / 3) << 4, nt = (t % 3) << 4;
    v8f acc = {};
#pragma unroll
    for (int k0 = 0; k0 < SPAD; k0 += 32)
      acc = wmma_f16(load_a16(ph, mt, k0, SPAD, lane),
                     load_b16(vT, nt, k0, SPAD, lane), acc);
    int n  = nt + (lane & 15);
    int mb = mt + ((lane >> 4) << 3);
#pragma unroll
    for (int r = 0; r < 8; ++r)
      o1h[(mb + r) * OPAD + n] = (_Float16)acc[r];
  }
  __syncthreads();

  // ---------------- Phase 4: OUT = O1 @ Wo^T + bo -> outf (f32) ---------------
  for (int t = wave; t < 12; t += 8) {
    int mt = (t / 3) << 4, nt = (t % 3) << 4;
    v8f acc = {};
#pragma unroll
    for (int k0 = 0; k0 < UPAD; k0 += 32)
      acc = wmma_f16(load_a16(o1h, mt, k0, OPAD, lane),
                     load_b16(woh, nt, k0, UPAD, lane), acc);
    int n  = nt + (lane & 15);
    int mb = mt + ((lane >> 4) << 3);
    if (n < UNIT) {
      float bov = bo[n];
#pragma unroll
      for (int r = 0; r < 8; ++r) {
        int s = mb + r;
        if (s < SEQ) outf[s * 48 + n] = acc[r] + bov;
      }
    }
  }
  __syncthreads();

  // ---------------- Phase 5: gate logits, softmax over s, weighted sum --------
  if (tid < SEQ) {
    float d = bq[0];
    for (int u = 0; u < UNIT; ++u) d += outf[tid * 48 + u] * Wq[u];
    float pm = mask[(size_t)b * SEQ + tid];
    sProb[tid] = d * pm - (1.f - pm) * NEG_BIG;
  }
  __syncthreads();
  if (tid == 0) {
    float mx = -3.4e38f;
    for (int s = 0; s < SEQ; ++s) mx = fmaxf(mx, sProb[s]);
    float sum = 0.f;
    for (int s = 0; s < SEQ; ++s) sum += __expf(sProb[s] - mx);
    sRed[0] = mx;
    sRed[1] = fast_rcp(sum);
  }
  __syncthreads();
  if (tid < UNIT) {
    float mx = sRed[0], inv = sRed[1];
    float acc = 0.f;
    for (int s = 0; s < SEQ; ++s)
      acc += outf[s * 48 + tid] * (__expf(sProb[s] - mx) * inv);
    out[(size_t)b * UNIT + tid] = acc;
  }
}

extern "C" void kernel_launch(void* const* d_in, const int* in_sizes, int n_in,
                              void* d_out, int out_size, void* d_ws, size_t ws_size,
                              hipStream_t stream) {
  const float* x        = (const float*)d_in[0];
  const float* mask     = (const float*)d_in[1];
  const float* Wi       = (const float*)d_in[2];
  const float* bi       = (const float*)d_in[3];
  const float* Wo       = (const float*)d_in[4];
  const float* bo       = (const float*)d_in[5];
  const float* scale_q  = (const float*)d_in[6];
  const float* offset_q = (const float*)d_in[7];
  const float* scale_k  = (const float*)d_in[8];
  const float* offset_k = (const float*)d_in[9];
  const float* Wq       = (const float*)d_in[10];
  const float* bq       = (const float*)d_in[11];
  float* out = (float*)d_out;

  const int B = in_sizes[0] / (SEQ * UNIT);   // 32768
  (void)n_in; (void)out_size; (void)d_ws; (void)ws_size;

  gau_kernel<<<B, 256, 0, stream>>>(x, mask, Wi, bi, Wo, bo,
                                    scale_q, offset_q, scale_k, offset_k,
                                    Wq, bq, out);
}
